// Attention_73452530696586
// MI455X (gfx1250) — matrix operand order repaired and checked by reading, not compiled, not run on hardware
//
#include <hip/hip_runtime.h>

// ---------------------------------------------------------------------------
// Types for CDNA5 WMMA (wave32): v_wmma_f32_16x16x32_bf16
// ---------------------------------------------------------------------------
typedef __attribute__((ext_vector_type(16))) __bf16 v16bf;
typedef __attribute__((ext_vector_type(8)))  float  v8f;

union Frag {              // 16 bf16 = 32 bytes = 8 VGPRs, loaded as 2x 16B
    v16bf v;
    uint4 q[2];
};

__device__ __forceinline__ unsigned short f2bf(float f) {
    unsigned int u = __float_as_uint(f);
    u += 0x7FFFu + ((u >> 16) & 1u);      // round-to-nearest-even
    return (unsigned short)(u >> 16);
}

__device__ __forceinline__ v8f zero8() {
    v8f z;
#pragma unroll
    for (int i = 0; i < 8; ++i) z[i] = 0.0f;
    return z;
}

// LDS byte offset of a __shared__ object: low 32 bits of the generic pointer
__device__ __forceinline__ unsigned lds_off(const void* p) {
    return (unsigned)(uintptr_t)p;
}

// Async bulk copy: 16 bytes global -> LDS, tracked by ASYNCcnt (no VGPR data).
__device__ __forceinline__ void async_cp16(unsigned ldsOff, const void* gaddr) {
    asm volatile("global_load_async_to_lds_b128 %0, %1, off"
                 :: "v"(ldsOff), "v"((unsigned long long)(uintptr_t)gaddr)
                 : "memory");
}
__device__ __forceinline__ void wait_async0() {
    asm volatile("s_wait_asynccnt 0x0" ::: "memory");
}

// ---------------------------------------------------------------------------
// Problem constants
// ---------------------------------------------------------------------------
static constexpr int B_    = 2;
static constexpr int S_    = 2048;
static constexpr int HID   = 4096;
static constexpr int NH    = 32;
static constexpr int NKV   = 8;
static constexpr int HD    = 128;
static constexpr int QKVD  = (NH + 2 * NKV) * HD;     // 6144
static constexpr int ROWS  = B_ * S_;                 // 4096

// Workspace layout (bytes). QB/KB/VBT alias the dead w_qkv_bf16 region after
// GEMM1; attn output aliases the dead x_bf16 region. Stream order makes this safe.
static constexpr size_t OFF_XB    = 0;                         // 33,554,432
static constexpr size_t OFF_ATTNB = 0;                         // alias (after GEMM1)
static constexpr size_t OFF_WQKVB = 33554432;                  // 50,331,648 (transposed)
static constexpr size_t OFF_QB    = 33554432;                  // alias (after GEMM1)
static constexpr size_t OFF_KB    = 67108864;
static constexpr size_t OFF_VBT   = 75497472;
static constexpr size_t OFF_WOUTB = 83886080;                  // 33,554,432 (transposed)
static constexpr size_t OFF_QKV32 = 117440512;                 // 100,663,296 (f32)
// total ws needed: 218,103,808 bytes

// ---------------------------------------------------------------------------
// fp32 -> bf16 raw conversion (row-major copy)
// ---------------------------------------------------------------------------
__global__ __launch_bounds__(256) void k_cvt_bf16(const float* __restrict__ src,
                                                  unsigned short* __restrict__ dst,
                                                  int n) {
    int i = blockIdx.x * 256 + threadIdx.x;
    if (i < n) dst[i] = f2bf(src[i]);
}

// ---------------------------------------------------------------------------
// fp32 [K][N] -> bf16 transposed [N][K], 32x32 LDS tiles (coalesced both sides).
// One-time pass so the GEMM hot loop stages B with pure b128 bulk copies.
// ---------------------------------------------------------------------------
__global__ __launch_bounds__(256) void k_cvt_bf16_t(const float* __restrict__ src,
                                                    unsigned short* __restrict__ dst,
                                                    int K, int N) {
    __shared__ float tile[32][33];
    const int n0 = blockIdx.x * 32;
    const int k0 = blockIdx.y * 32;
    const int tx = threadIdx.x & 31;
    const int ty = threadIdx.x >> 5;          // 0..7
#pragma unroll
    for (int r = 0; r < 4; ++r)
        tile[ty + 8 * r][tx] = src[(size_t)(k0 + ty + 8 * r) * N + n0 + tx];
    __syncthreads();
#pragma unroll
    for (int r = 0; r < 4; ++r)
        dst[(size_t)(n0 + ty + 8 * r) * K + k0 + tx] = f2bf(tile[tx][ty + 8 * r]);
}

// ---------------------------------------------------------------------------
// RoPE + split: qkv fp32 [ROWS][48][128] ->
//   qb  bf16 [b][h][s][d]      (roped)
//   kb  bf16 [b][hkv][s][d]    (roped)
//   vbT bf16 [b][hkv][d][s]    (transposed for contiguous P.V B-fragments)
// ---------------------------------------------------------------------------
__global__ __launch_bounds__(256) void k_rope_split(const float* __restrict__ qkv,
                                                    unsigned short* __restrict__ qb,
                                                    unsigned short* __restrict__ kb,
                                                    unsigned short* __restrict__ vbT) {
    int i = blockIdx.x * 256 + threadIdx.x;     // < B*S*48*64
    int d    = i & 63;
    int head = (i >> 6) % 48;
    int s    = (i / (64 * 48)) % S_;
    int b    = i / (64 * 48 * S_);

    size_t base = ((size_t)(b * S_ + s) * 48 + head) * HD;
    float t1 = qkv[base + d];
    float t2 = qkv[base + 64 + d];

    float o1 = t1, o2 = t2;
    if (head < NH + NKV) {   // rope applies to q and k only
        float inv = __powf(10000.0f, -(float)d * (1.0f / 64.0f));
        float f   = (float)s * inv;
        float c, sn;
        __sincosf(f, &sn, &c);
        o1 = t1 * c - t2 * sn;
        o2 = t1 * sn + t2 * c;
    }

    if (head < NH) {
        size_t o = ((size_t)(b * NH + head) * S_ + s) * HD;
        qb[o + d]      = f2bf(o1);
        qb[o + 64 + d] = f2bf(o2);
    } else if (head < NH + NKV) {
        int hk = head - NH;
        size_t o = ((size_t)(b * NKV + hk) * S_ + s) * HD;
        kb[o + d]      = f2bf(o1);
        kb[o + 64 + d] = f2bf(o2);
    } else {
        int hk = head - NH - NKV;
        size_t o = ((size_t)(b * NKV + hk) * HD) * S_;
        vbT[o + (size_t)d * S_ + s]        = f2bf(t1);
        vbT[o + (size_t)(64 + d) * S_ + s] = f2bf(t2);
    }
}

// ---------------------------------------------------------------------------
// bf16 GEMM, fp32 out:  C[M,N] = A[M,K] @ Bt[N,K]^T   (raw bf16, Bt transposed)
// Block tile 128x128, 8 waves, wave tile 32x64 (2x4 WMMA accumulators), K-step 32.
// Double-buffered LDS fed by GLOBAL_LOAD_ASYNC_TO_LDS_B128 (ASYNCcnt).
// Both LDS tiles are [row][k] so every fragment is two 16B reads from one row:
//   A frag: VGPR i <-> K = 8*khalf + {0..7, 16..23}, row = lane%16
//   B frag: VGPR i <-> K = 8*khalf + {0..7, 16..23}, col = lane%16
// ---------------------------------------------------------------------------
__global__ __launch_bounds__(256) void k_gemm_bf16(const unsigned short* __restrict__ A,
                                                   const unsigned short* __restrict__ Bt,
                                                   float* __restrict__ C,
                                                   int M, int N, int K) {
    __shared__ __align__(16) unsigned short lA[2][128 * 32];
    __shared__ __align__(16) unsigned short lB[2][128 * 32];

    const int tid   = threadIdx.x;
    const int wave  = tid >> 5;
    const int lane  = tid & 31;
    const int lm    = lane & 15;
    const int khalf = lane >> 4;
    const int half8 = khalf * 8;
    const int wm    = wave & 3;        // 4 waves along M (4*32 = 128)
    const int wn    = wave >> 2;       // 2 waves along N (2*64 = 128)
    const int mBlk  = blockIdx.y * 128;
    const int nBlk  = blockIdx.x * 128;

    // staging: 256 threads x 32B (2x async b128) per matrix per K-step
    const int srow = tid >> 1;             // 128 rows, 2 threads per row
    const int skq  = (tid & 1) * 16;       // 16 bf16 each

    auto stage = [&](int buf, int k0) {
        const unsigned short* ga = A  + (size_t)(mBlk + srow) * K + k0 + skq;
        const unsigned short* gb = Bt + (size_t)(nBlk + srow) * K + k0 + skq;
        unsigned oa = lds_off(&lA[buf][srow * 32 + skq]);
        unsigned ob = lds_off(&lB[buf][srow * 32 + skq]);
        async_cp16(oa,      ga);
        async_cp16(oa + 16, ga + 8);
        async_cp16(ob,      gb);
        async_cp16(ob + 16, gb + 8);
    };

    v8f acc[2][4];
#pragma unroll
    for (int ms = 0; ms < 2; ++ms)
#pragma unroll
        for (int ns = 0; ns < 4; ++ns) acc[ms][ns] = zero8();

    stage(0, 0);
    int cur = 0;
    for (int k0 = 0; k0 < K; k0 += 32) {
        wait_async0();          // own wave's async writes to `cur` done
        __syncthreads();        // everyone's are done
        if (k0 + 32 < K) stage(cur ^ 1, k0 + 32);   // overlap next tile

        Frag fa[2], fb[4];
#pragma unroll
        for (int ms = 0; ms < 2; ++ms) {
            const unsigned short* p = &lA[cur][(wm * 32 + ms * 16 + lm) * 32];
            fa[ms].q[0] = *(const uint4*)(p + half8);        // K = 8*khalf .. +7
            fa[ms].q[1] = *(const uint4*)(p + 16 + half8);   // K = 16+8*khalf ..
        }
#pragma unroll
        for (int ns = 0; ns < 4; ++ns) {
            const unsigned short* p =
                &lB[cur][(wn * 64 + ns * 16 + lm) * 32];
            fb[ns].q[0] = *(const uint4*)(p + half8);
            fb[ns].q[1] = *(const uint4*)(p + 16 + half8);
        }
#pragma unroll
        for (int ms = 0; ms < 2; ++ms)
#pragma unroll
            for (int ns = 0; ns < 4; ++ns)
                acc[ms][ns] = __builtin_amdgcn_wmma_f32_16x16x32_bf16(
                    false, fa[ms].v, false, fb[ns].v, (short)0, acc[ms][ns],
                    false, false);
        cur ^= 1;
    }

    // ---- epilogue: C layout VGPR i <-> row i + 8*khalf, col = lane%16 ----
#pragma unroll
    for (int ms = 0; ms < 2; ++ms) {
        int rbase = mBlk + wm * 32 + ms * 16 + half8;
#pragma unroll
        for (int ns = 0; ns < 4; ++ns) {
            int col = nBlk + wn * 64 + ns * 16 + lm;
#pragma unroll
            for (int i = 0; i < 8; ++i)
                C[(size_t)(rbase + i) * N + col] = acc[ms][ns][i];
        }
    }
}

// ---------------------------------------------------------------------------
// Flash attention (non-causal, GQA group=4), bf16 WMMA, fp32 accumulate.
// Grid (S/64, NH, B), block 128 (4 waves). Each wave: 16 q rows, full D=128.
// K/V tiles for each 32-key chunk are shared by all 4 waves -> staged ONCE per
// block into double-buffered LDS via async copies (4x less vmem traffic).
// ---------------------------------------------------------------------------
__global__ __launch_bounds__(128) void k_attn(const unsigned short* __restrict__ qb,
                                              const unsigned short* __restrict__ kb,
                                              const unsigned short* __restrict__ vbT,
                                              unsigned short* __restrict__ attnb) {
    __shared__ __align__(16) unsigned short lK[2][32 * 128];   // [key][d]   8KB x2
    __shared__ __align__(16) unsigned short lV[2][128 * 32];   // [d][key]   8KB x2
    __shared__ __align__(16) unsigned short lP[4][16 * 32];    // per-wave P tile

    const int b  = blockIdx.z;
    const int h  = blockIdx.y;
    const int qt = blockIdx.x;
    const int tid   = threadIdx.x;
    const int wave  = tid >> 5;
    const int lane  = tid & 31;
    const int lm    = lane & 15;
    const int khalf = lane >> 4;
    const int half8 = khalf * 8;
    const int hkv = h >> 2;                       // group = NH/NKV = 4
    const int q0  = qt * 64 + wave * 16;

    const unsigned short* kbase = kb  + (size_t)(b * NKV + hkv) * S_ * HD;
    const unsigned short* vbase = vbT + (size_t)(b * NKV + hkv) * HD * S_;

    // staging (128 threads): K chunk 32x128, V chunk 128x32, 6 async b128 each
    const int krow = tid >> 2;            // 0..31  (key row)
    const int kdq  = (tid & 3) * 32;      // d offset
    auto stage = [&](int buf, int kv0) {
        const unsigned short* gk = kbase + (size_t)(kv0 + krow) * HD + kdq;
        unsigned ok = lds_off(&lK[buf][krow * 128 + kdq]);
        async_cp16(ok,      gk);
        async_cp16(ok + 16, gk + 8);
        const unsigned short* gv = vbase + (size_t)tid * S_ + kv0;   // d row = tid
        unsigned ov = lds_off(&lV[buf][tid * 32]);
#pragma unroll
        for (int j = 0; j < 4; ++j) async_cp16(ov + 16 * j, gv + 8 * j);
    };

    // Preload Q A-fragments for all 4 D-chunks (lane row = q0 + lm)
    const unsigned short* qrow =
        qb + ((size_t)((b * NH + h) * S_) + q0 + lm) * HD;
    Frag fq[4];
#pragma unroll
    for (int dc = 0; dc < 4; ++dc) {
        fq[dc].q[0] = *(const uint4*)(qrow + dc * 32 + half8);
        fq[dc].q[1] = *(const uint4*)(qrow + dc * 32 + 16 + half8);
    }

    v8f o[8];
#pragma unroll
    for (int t = 0; t < 8; ++t) o[t] = zero8();
    float rowm[8], rowl[8];
#pragma unroll
    for (int i = 0; i < 8; ++i) { rowm[i] = -1e30f; rowl[i] = 0.0f; }

    const float scl = 0.08838834764831845f * 1.4426950408889634f; // D^-0.5 * log2e

    stage(0, 0);
    int cur = 0;
    for (int kv0 = 0; kv0 < S_; kv0 += 32) {
        wait_async0();
        __syncthreads();
        if (kv0 + 32 < S_) stage(cur ^ 1, kv0 + 32);

        // ---- scores: two 16x16 tiles over D=128 (4 chunks of 32) ----
        v8f s0 = zero8(), s1 = zero8();
#pragma unroll
        for (int dc = 0; dc < 4; ++dc) {
            const unsigned short* kr = &lK[cur][lm * 128 + dc * 32];
            Frag fk;
            fk.q[0] = *(const uint4*)(kr + half8);
            fk.q[1] = *(const uint4*)(kr + 16 + half8);
            s0 = __builtin_amdgcn_wmma_f32_16x16x32_bf16(
                false, fq[dc].v, false, fk.v, (short)0, s0, false, false);
        }
#pragma unroll
        for (int dc = 0; dc < 4; ++dc) {
            const unsigned short* kr =
                &lK[cur][(16 + lm) * 128 + dc * 32];
            Frag fk;
            fk.q[0] = *(const uint4*)(kr + half8);
            fk.q[1] = *(const uint4*)(kr + 16 + half8);
            s1 = __builtin_amdgcn_wmma_f32_16x16x32_bf16(
                false, fq[dc].v, false, fk.v, (short)0, s1, false, false);
        }

        // ---- online softmax over the 32 new columns (half-wave reductions) ----
#pragma unroll
        for (int i = 0; i < 8; ++i) {
            float a0 = s0[i] * scl;
            float a1 = s1[i] * scl;
            float t = fmaxf(a0, a1);
#pragma unroll
            for (int off = 8; off >= 1; off >>= 1)
                t = fmaxf(t, __shfl_xor(t, off, 16));
            float nm   = fmaxf(rowm[i], t);
            float corr = exp2f(rowm[i] - nm);
            float p0   = exp2f(a0 - nm);
            float p1   = exp2f(a1 - nm);
            float ps   = p0 + p1;
#pragma unroll
            for (int off = 8; off >= 1; off >>= 1)
                ps += __shfl_xor(ps, off, 16);
            rowl[i] = rowl[i] * corr + ps;
            rowm[i] = nm;
#pragma unroll
            for (int t8 = 0; t8 < 8; ++t8) o[t8][i] *= corr;
            // C-layout -> A-layout reshape via LDS (row = i + 8*khalf, k = col)
            lP[wave][(i + half8) * 32 + lm]      = f2bf(p0);
            lP[wave][(i + half8) * 32 + 16 + lm] = f2bf(p1);
        }
        asm volatile("s_wait_dscnt 0x0" ::: "memory");   // same-wave LDS RAW

        // ---- P (16x32) @ V (32x128): 8 WMMAs ----
        Frag fp;
        const unsigned short* pb = &lP[wave][lm * 32];
        fp.q[0] = *(const uint4*)(pb + half8);
        fp.q[1] = *(const uint4*)(pb + 16 + half8);
#pragma unroll
        for (int t = 0; t < 8; ++t) {
            const unsigned short* vr = &lV[cur][(t * 16 + lm) * 32];
            Frag fv;
            fv.q[0] = *(const uint4*)(vr + half8);
            fv.q[1] = *(const uint4*)(vr + 16 + half8);
            o[t] = __builtin_amdgcn_wmma_f32_16x16x32_bf16(
                false, fp.v, false, fv.v, (short)0, o[t], false, false);
        }
        cur ^= 1;
    }

    // ---- normalize and store bf16 [row = b*S+q, col = h*128 + d] ----
#pragma unroll
    for (int i = 0; i < 8; ++i) {
        float inv = 1.0f / rowl[i];
        int qg = q0 + i + half8;
        size_t rbase = (size_t)(b * S_ + qg) * (NH * HD) + h * HD;
#pragma unroll
        for (int t = 0; t < 8; ++t)
            attnb[rbase + t * 16 + lm] = f2bf(o[t][i] * inv);
    }
}

// ---------------------------------------------------------------------------
// Host-side launcher
// ---------------------------------------------------------------------------
extern "C" void kernel_launch(void* const* d_in, const int* in_sizes, int n_in,
                              void* d_out, int out_size, void* d_ws, size_t ws_size,
                              hipStream_t stream) {
    const float* x     = (const float*)d_in[0];   // [2,2048,4096]
    const float* w_qkv = (const float*)d_in[1];   // [4096,6144]
    const float* w_out = (const float*)d_in[2];   // [4096,4096]
    float* out = (float*)d_out;                   // [2,2048,4096]

    unsigned char* ws = (unsigned char*)d_ws;
    unsigned short* xb     = (unsigned short*)(ws + OFF_XB);
    unsigned short* wqkvbT = (unsigned short*)(ws + OFF_WQKVB);   // [6144][4096]
    unsigned short* woutbT = (unsigned short*)(ws + OFF_WOUTB);   // [4096][4096]
    float*          qkv32  = (float*)         (ws + OFF_QKV32);
    unsigned short* qb     = (unsigned short*)(ws + OFF_QB);
    unsigned short* kbuf   = (unsigned short*)(ws + OFF_KB);
    unsigned short* vbt    = (unsigned short*)(ws + OFF_VBT);
    unsigned short* attnb  = (unsigned short*)(ws + OFF_ATTNB);

    const int n_x = ROWS * HID;     // 16,777,216

    k_cvt_bf16<<<n_x / 256, 256, 0, stream>>>(x, xb, n_x);
    // weights converted AND transposed once -> GEMM stages B as bulk copies
    k_cvt_bf16_t<<<dim3(QKVD / 32, HID / 32), 256, 0, stream>>>(w_qkv, wqkvbT, HID, QKVD);
    k_cvt_bf16_t<<<dim3(HID / 32, HID / 32), 256, 0, stream>>>(w_out, woutbT, HID, HID);

    // qkv = x @ w_qkv   (4096 x 6144, K=4096) -> fp32
    k_gemm_bf16<<<dim3(QKVD / 128, ROWS / 128), 256, 0, stream>>>(
        xb, wqkvbT, qkv32, ROWS, QKVD, HID);

    // rope + split (overwrites dead w_qkv_bf16 region)
    const int n_pairs = B_ * S_ * 48 * 64;   // 12,582,912
    k_rope_split<<<n_pairs / 256, 256, 0, stream>>>(qkv32, qb, kbuf, vbt);

    // flash attention (overwrites dead x_bf16 region)
    k_attn<<<dim3(S_ / 64, NH, B_), 128, 0, stream>>>(qb, kbuf, vbt, attnb);

    // out = attn @ w_out  (4096 x 4096, K=4096) -> fp32 d_out
    k_gemm_bf16<<<dim3(HID / 128, ROWS / 128), 256, 0, stream>>>(
        attnb, woutbT, out, ROWS, HID, HID);
}
